// DynamicsModel_75574244540868
// MI455X (gfx1250) — matrix operand order repaired and checked by reading, not compiled, not run on hardware
//
#include <hip/hip_runtime.h>
#include <hip/hip_bf16.h>

// ---------------------------------------------------------------------------
// 2-layer LSTM (B=512, T=1024, F=128, H=512) + binarize + FC(512->32).
// Batch-parallel recurrence: each workgroup owns 32 batch rows (TWO WMMA
// M-tiles) and runs all 1024 timesteps locally. Two M-tiles per B-tile halves
// the dominant cost: L2->WGP weight streaming (each wmma B operand is 1KB,
// reused across both A tiles). B tiles for all 4 gates are batch-loaded into
// separate registers per k-block so each s_wait covers 8 WMMAs (deep
// load->wmma pipelining instead of wait-0 per gate).
//   - h0/h1 state: LDS bf16, bank-staggered stride (conflict-free b128 loads)
//   - c0/c1 state: registers (f32 WMMA C/D layout maps 1:1 to lanes/regs)
//   - weights: converted once to bf16 in workspace; L2-resident (6.5 MB)
//   - GEMMs: v_wmma_f32_16x16x32_bf16; 4 gates x 2 M-tiles share each A/B
// ---------------------------------------------------------------------------

typedef __attribute__((ext_vector_type(16))) __bf16 v16bf;
typedef __attribute__((ext_vector_type(8)))  float  v8f;

#define B_SZ   512
#define T_SZ   1024
#define F_SZ   128
#define H_SZ   512
#define OUT_SZ 32
#define ROWS   32      // batch rows per workgroup (two WMMA M tiles)
#define HSTR   520     // LDS row stride for h buffers (bf16 elems), bank-staggered
#define XSTR   136     // LDS row stride for x tile

union V16 { v16bf v; uint4 q[2]; };

__device__ __forceinline__ float sigmoidf_(float x) {
  return 1.0f / (1.0f + __expf(-x));
}

// acc[0..3] = gate accumulators for M-tile 0, acc[4..7] for M-tile 1.
// A VGPR layout (16-bit, 16x32): lanes 0-15 hold K 0..15 (two b128),
// lanes 16-31 hold K 16..31. B layout (32x16): lanes 0-15 hold
// K 0..15 of W row n (32 contiguous bytes), lanes 16-31 hold K 16..31.
template <int KDIM>
__device__ __forceinline__ void gemm_gates2(v8f acc[8],
                                            const __bf16* __restrict__ W,
                                            int j0,
                                            const __bf16* __restrict__ A,
                                            int astride, int lm, int hf) {
  const __bf16* arow0 = A + lm * astride;
  const __bf16* arow1 = A + (16 + lm) * astride;
  const int kA = hf * 16;   // A q0 K offset for this lane half
  const int kB = hf * 16;   // B K offset for this lane half
#pragma unroll 2
  for (int kblk = 0; kblk < KDIM; kblk += 32) {
    // ---- batch all loads for this k-block into one issue window ----------
    V16 a0, a1, b[4];
    a0.q[0] = *(const uint4*)(arow0 + kblk + kA);        // ds_load_b128
    a0.q[1] = *(const uint4*)(arow0 + kblk + kA + 8);
    a1.q[0] = *(const uint4*)(arow1 + kblk + kA);
    a1.q[1] = *(const uint4*)(arow1 + kblk + kA + 8);
#pragma unroll
    for (int g = 0; g < 4; ++g) {
      const int n = g * H_SZ + j0 + lm;                  // gate column (W row)
      const uint4* pb = (const uint4*)(W + (size_t)n * KDIM + kblk + kB);
      b[g].q[0] = pb[0];                                 // global_load_b128
      b[g].q[1] = pb[1];
    }
    // ---- 8 WMMAs consume the batched tiles -------------------------------
#pragma unroll
    for (int g = 0; g < 4; ++g) {
      acc[g] = __builtin_amdgcn_wmma_f32_16x16x32_bf16(
          false, a0.v, false, b[g].v, (short)0, acc[g], false, false);
      acc[4 + g] = __builtin_amdgcn_wmma_f32_16x16x32_bf16(
          false, a1.v, false, b[g].v, (short)0, acc[4 + g], false, false);
    }
  }
}

__global__ void lstm2_cvt_bf16_kernel(const float* __restrict__ src,
                                      __bf16* __restrict__ dst, int n) {
  int i = blockIdx.x * blockDim.x + threadIdx.x;
  if (i < n) dst[i] = (__bf16)src[i];
}

__global__ __launch_bounds__(1024, 1)
void lstm2_recurrence_kernel(const float* __restrict__ x,
                             const float* __restrict__ bih0,
                             const float* __restrict__ bhh0,
                             const float* __restrict__ bih1,
                             const float* __restrict__ bhh1,
                             const float* __restrict__ Wfc,
                             const float* __restrict__ bfc,
                             const __bf16* __restrict__ Wih0,
                             const __bf16* __restrict__ Whh0,
                             const __bf16* __restrict__ Wih1,
                             const __bf16* __restrict__ Whh1,
                             float* __restrict__ out) {
  __shared__ __bf16 h0buf[ROWS * HSTR];
  __shared__ __bf16 h1buf[ROWS * HSTR];
  __shared__ __bf16 xbuf [ROWS * XSTR];

  const int tid  = threadIdx.x;
  const int wg   = blockIdx.x;        // 0..15 : batch tile (32 rows)
  const int wv   = tid >> 5;          // wave 0..31 : owns 16 hidden columns
  const int lane = tid & 31;
  const int lm   = lane & 15;
  const int hf   = lane >> 4;
  const int j0   = wv * 16;           // hidden columns [j0, j0+16)
  const int brow0 = wg * ROWS;        // first batch row of this WG

  // zero-init recurrent state
  for (int i = tid; i < ROWS * HSTR; i += 1024) {
    h0buf[i] = (__bf16)0.0f;
    h1buf[i] = (__bf16)0.0f;
  }
  float c0r[16], c1r[16];             // [mtile*8 + r]
#pragma unroll
  for (int r = 0; r < 16; ++r) { c0r[r] = 0.0f; c1r[r] = 0.0f; }

  // combined biases for this lane's gate columns (same for all C rows)
  float bias0[4], bias1[4];
#pragma unroll
  for (int g = 0; g < 4; ++g) {
    const int col = g * H_SZ + j0 + lm;
    bias0[g] = bih0[col] + bhh0[col];
    bias1[g] = bih1[col] + bhh1[col];
  }
  __syncthreads();

  for (int t = 0; t < T_SZ; ++t) {
    // ---- stage x_t tile [32,128] as bf16 into LDS -----------------------
    for (int i = tid; i < ROWS * F_SZ; i += 1024) {
      const int r = i >> 7;          // /F_SZ
      const int f = i & (F_SZ - 1);
      xbuf[r * XSTR + f] =
          (__bf16)x[((size_t)(brow0 + r) * T_SZ + t) * F_SZ + f];
    }
    // prefetch next timestep's x tile (global_prefetch_b8), 128B granules
    if (t + 1 < T_SZ && tid < 128) {
      const int r = tid >> 2;
      const int f = (tid & 3) * 32;
      __builtin_prefetch(&x[((size_t)(brow0 + r) * T_SZ + (t + 1)) * F_SZ + f], 0, 1);
    }
    __syncthreads();

    // ---- layer 0: gates = x@Wih0^T + h0@Whh0^T + b ----------------------
    v8f acc[8];
#pragma unroll
    for (int g = 0; g < 4; ++g)
#pragma unroll
      for (int r = 0; r < 8; ++r) { acc[g][r] = bias0[g]; acc[4 + g][r] = bias0[g]; }

    gemm_gates2<F_SZ>(acc, Wih0, j0, xbuf,  XSTR, lm, hf);
    gemm_gates2<H_SZ>(acc, Whh0, j0, h0buf, HSTR, lm, hf);
    __syncthreads();   // all waves done reading old h0

#pragma unroll
    for (int mt = 0; mt < 2; ++mt) {
#pragma unroll
      for (int r = 0; r < 8; ++r) {
        const float ig = sigmoidf_(acc[4 * mt + 0][r]);
        const float fg = sigmoidf_(acc[4 * mt + 1][r]);
        const float gg = tanhf(acc[4 * mt + 2][r]);
        const float og = sigmoidf_(acc[4 * mt + 3][r]);
        const float cn = fg * c0r[8 * mt + r] + ig * gg;
        c0r[8 * mt + r] = cn;
        const int m = 16 * mt + r + hf * 8;   // C/D layout row
        h0buf[m * HSTR + j0 + lm] = (__bf16)(og * tanhf(cn));
      }
    }
    __syncthreads();   // new h0 visible

    // ---- layer 1: gates = h0@Wih1^T + h1@Whh1^T + b ---------------------
#pragma unroll
    for (int g = 0; g < 4; ++g)
#pragma unroll
      for (int r = 0; r < 8; ++r) { acc[g][r] = bias1[g]; acc[4 + g][r] = bias1[g]; }

    gemm_gates2<H_SZ>(acc, Wih1, j0, h0buf, HSTR, lm, hf);
    gemm_gates2<H_SZ>(acc, Whh1, j0, h1buf, HSTR, lm, hf);
    __syncthreads();   // all waves done reading old h1

#pragma unroll
    for (int mt = 0; mt < 2; ++mt) {
#pragma unroll
      for (int r = 0; r < 8; ++r) {
        const float ig = sigmoidf_(acc[4 * mt + 0][r]);
        const float fg = sigmoidf_(acc[4 * mt + 1][r]);
        const float gg = tanhf(acc[4 * mt + 2][r]);
        const float og = sigmoidf_(acc[4 * mt + 3][r]);
        const float cn = fg * c1r[8 * mt + r] + ig * gg;
        c1r[8 * mt + r] = cn;
        const int m = 16 * mt + r + hf * 8;
        h1buf[m * HSTR + j0 + lm] = (__bf16)(og * tanhf(cn));
      }
    }
    __syncthreads();   // new h1 visible
  }

  // ---- epilogue: out = binarize(h1_last) @ Wfc^T + bfc  [32 x 32] -------
  {
    const int r = tid >> 5;           // batch row within tile (0..31)
    const int o = tid & (OUT_SZ - 1); // output column
    float s = bfc[o];
    const __bf16* hrow = h1buf + r * HSTR;
    const float* wrow = Wfc + o * H_SZ;
#pragma unroll 8
    for (int j = 0; j < H_SZ; ++j) {
      if ((float)hrow[j] > 0.0f) s += wrow[j];
    }
    out[(size_t)(brow0 + r) * OUT_SZ + o] = s;
  }
}

extern "C" void kernel_launch(void* const* d_in, const int* in_sizes, int n_in,
                              void* d_out, int out_size, void* d_ws, size_t ws_size,
                              hipStream_t stream) {
  const float* x    = (const float*)d_in[0];
  const float* Wih0 = (const float*)d_in[1];   // [2048,128]
  const float* Whh0 = (const float*)d_in[2];   // [2048,512]
  const float* bih0 = (const float*)d_in[3];
  const float* bhh0 = (const float*)d_in[4];
  const float* Wih1 = (const float*)d_in[5];   // [2048,512]
  const float* Whh1 = (const float*)d_in[6];   // [2048,512]
  const float* bih1 = (const float*)d_in[7];
  const float* bhh1 = (const float*)d_in[8];
  const float* Wfc  = (const float*)d_in[9];   // [32,512]
  const float* bfc  = (const float*)d_in[10];
  float* out = (float*)d_out;

  // bf16 weight staging in workspace (6.5 MB)
  __bf16* wih0 = (__bf16*)d_ws;
  __bf16* whh0 = wih0 + 2048 * 128;
  __bf16* wih1 = whh0 + 2048 * 512;
  __bf16* whh1 = wih1 + 2048 * 512;

  const int n_ih0 = 2048 * 128;
  const int n_hh  = 2048 * 512;
  lstm2_cvt_bf16_kernel<<<(n_ih0 + 255) / 256, 256, 0, stream>>>(Wih0, wih0, n_ih0);
  lstm2_cvt_bf16_kernel<<<(n_hh  + 255) / 256, 256, 0, stream>>>(Whh0, whh0, n_hh);
  lstm2_cvt_bf16_kernel<<<(n_hh  + 255) / 256, 256, 0, stream>>>(Wih1, wih1, n_hh);
  lstm2_cvt_bf16_kernel<<<(n_hh  + 255) / 256, 256, 0, stream>>>(Whh1, whh1, n_hh);

  lstm2_recurrence_kernel<<<B_SZ / ROWS, 1024, 0, stream>>>(
      x, bih0, bhh0, bih1, bhh1, Wfc, bfc, wih0, whh0, wih1, whh1, out);
}
